// TwoStageEncoderLayer_90675349553302
// MI455X (gfx1250) — matrix operand order repaired and checked by reading, not compiled, not run on hardware
//
#include <hip/hip_runtime.h>
#include <math.h>

#define B_   4
#define C_   32
#define L_   96
#define D_   16
#define DM_  256
#define DFF_ 1024
#define H_   8
#define NE_  17
#define NF_  48
#define NTOK_MAIN (B_*C_)         // 128
#define NROWS16   (B_*C_*L_)      // 12288
#define XELEMS    (B_*C_*L_*D_)   // 196608

typedef __attribute__((ext_vector_type(16))) _Float16 v16h;
typedef __attribute__((ext_vector_type(8)))  float    v8f;
typedef __attribute__((ext_vector_type(8)))  _Float16 h8;

// ---------------------------------------------------------------------------
// WMMA GEMM: C[M,N] = A[M,K] @ W[K,N] (+bias[N]) (+pe[(m%pe_mod),N]) (relu?)
// 128 threads = 4 waves; 32x64 macro-tile, each wave owns a 16x32 strip
// (2 accumulators, shared A fragment). LDS layouts are shaped to the ISA
// fragment geometry so af/bf are single contiguous 32B loads:
//   As chunk layout: [(m*2+half)*16] = A cols half*16..+15 (contiguous k)
//   Bt transposed:   [n*32 + k]      = B col n, K-contiguous
// Ping-pong double buffering: one barrier per k-step, next tile's global
// loads issued before the WMMAs. Assumes M % 32 == 0, K % 16 == 0, K >= 32
// (true for every GEMM in this model).
// ---------------------------------------------------------------------------
__global__ __launch_bounds__(128)
void gemm_wmma(const float* __restrict__ A, const float* __restrict__ W,
               const float* __restrict__ bias, const float* __restrict__ pe, int pe_mod,
               float* __restrict__ Cout, int M, int N, int K, int relu,
               const int* __restrict__ active, int eid)
{
  if (active && active[eid] == 0) return;   // gated-expert early exit (uniform)
  __shared__ _Float16 AsB[2][32 * 32];
  __shared__ _Float16 BtB[2][64 * 32];
  const int tid   = threadIdx.x;
  const int lane  = tid & 31, wave = tid >> 5;
  const int half  = lane >> 4, mr = lane & 15;
  const int mhalf = wave >> 1, nq = wave & 1;
  const int m0 = blockIdx.y * 32, n0 = blockIdx.x * 64;
  const int lrA = tid >> 2, lcA = (tid & 3) * 8;    // A: row, 8-col run
  const int lrB = tid >> 2, cbB = (tid & 3) * 16;   // B: row, 16-col run
  const bool nfast = (n0 + 64 <= N) && ((N & 3) == 0);
  const float* apA = A + (size_t)(m0 + lrA) * K;    // always < M

  float aR[8]; bool aOk[2];
  float bR[16]; bool bOk = false;

  auto loadA = [&](int k0) {
    int rem = K - k0;                        // multiple of 16, >= 16
    if (k0 + 64 < K) __builtin_prefetch(apA + k0 + 64, 0, 1);
#pragma unroll
    for (int g = 0; g < 2; ++g) {
      int c = lcA + g * 4;
      bool ok = c < rem;                     // whole float4 group in/out
      aOk[g] = ok;
      float4 v4 = *(const float4*)(apA + k0 + (ok ? c : 0));
      aR[g*4+0] = v4.x; aR[g*4+1] = v4.y; aR[g*4+2] = v4.z; aR[g*4+3] = v4.w;
    }
  };
  auto loadB = [&](int k0) {
    int kk = k0 + lrB;
    bOk = kk < K;
    const float* wp = W + (size_t)(bOk ? kk : 0) * N;
    if (nfast) {
#pragma unroll
      for (int g = 0; g < 4; ++g) {
        float4 v4 = *(const float4*)(wp + n0 + cbB + g * 4);
        bR[g*4+0] = v4.x; bR[g*4+1] = v4.y; bR[g*4+2] = v4.z; bR[g*4+3] = v4.w;
      }
    } else {
#pragma unroll
      for (int e = 0; e < 16; ++e) {
        int nn = n0 + cbB + e;
        float v = wp[nn < N ? nn : (N - 1)];
        bR[e] = (nn < N) ? v : 0.f;
      }
    }
  };
  auto storeA = [&](int buf) {
    int dst = (lrA * 2 + (lcA >> 4)) * 16 + (lcA & 15);
    h8 hv;
#pragma unroll
    for (int j = 0; j < 8; ++j)
      hv[j] = aOk[j >> 2] ? (_Float16)aR[j] : (_Float16)0.f;
    *(h8*)&AsB[buf][dst] = hv;
  };
  auto storeB = [&](int buf) {
#pragma unroll
    for (int e = 0; e < 16; ++e)
      BtB[buf][(cbB + e) * 32 + lrB] = bOk ? (_Float16)bR[e] : (_Float16)0.f;
  };

  v8f accs[2];
  accs[0] = (v8f){0.f,0.f,0.f,0.f,0.f,0.f,0.f,0.f};
  accs[1] = accs[0];

  loadA(0); loadB(0);
  storeA(0); storeB(0);
  int cur = 0;
  for (int k0 = 0; k0 < K; k0 += 32) {
    __syncthreads();
    const bool more = (k0 + 32 < K);
    if (more) { loadA(k0 + 32); loadB(k0 + 32); }   // overlap with WMMAs below
    const _Float16* As = AsB[cur];
    const _Float16* Bt = BtB[cur];
    v16h af  = *(const v16h*)&As[((mhalf * 16 + mr) * 2 + half) * 16];
    v16h bf0 = *(const v16h*)&Bt[(nq * 32 + mr)      * 32 + (half << 4)];
    v16h bf1 = *(const v16h*)&Bt[(nq * 32 + 16 + mr) * 32 + (half << 4)];
    accs[0] = __builtin_amdgcn_wmma_f32_16x16x32_f16(false, af, false, bf0,
                                                     (short)0, accs[0], false, false);
    accs[1] = __builtin_amdgcn_wmma_f32_16x16x32_f16(false, af, false, bf1,
                                                     (short)0, accs[1], false, false);
    if (more) { storeA(cur ^ 1); storeB(cur ^ 1); }
    cur ^= 1;
  }
  // C/D layout: half 0 -> sub-rows 0..7, half 1 -> 8..15, col = lane&15
#pragma unroll
  for (int t = 0; t < 2; ++t) {
#pragma unroll
    for (int i = 0; i < 8; ++i) {
      int mm = m0 + mhalf * 16 + half * 8 + i;
      int nn = n0 + nq * 32 + t * 16 + mr;
      if (nn < N) {
        float v = accs[t][i];
        if (bias) v += bias[nn];
        if (pe)   v += pe[(size_t)(mm % pe_mod) * N + nn];
        if (relu) v = fmaxf(v, 0.f);
        Cout[(size_t)mm * N + nn] = v;
      }
    }
  }
}

// ---------------------------------------------------------------------------
// Fused softmax attention, one block per (sequence, head), one thread per query.
// S <= 48, E = 32. K/V staged in LDS, two-pass (max, then exp-sum + accumulate).
// ---------------------------------------------------------------------------
__global__ void attn_fused(const float* __restrict__ Q, const float* __restrict__ K,
                           const float* __restrict__ V, float* __restrict__ O,
                           int S, const int* __restrict__ active, int eid)
{
  if (active && active[eid] == 0) return;
  extern __shared__ float sm[];
  float* Ks = sm;
  float* Vs = sm + S * 32;
  const int n = blockIdx.x, h = blockIdx.y;
  const int l = threadIdx.x;
  const size_t base = (size_t)n * S * DM_ + h * 32;
  for (int t = l; t < S * 32; t += blockDim.x) {
    int s = t >> 5, d = t & 31;
    Ks[t] = K[base + (size_t)s * DM_ + d];
    Vs[t] = V[base + (size_t)s * DM_ + d];
  }
  __syncthreads();
  float qv[32];
#pragma unroll
  for (int d = 0; d < 32; ++d) qv[d] = Q[base + (size_t)l * DM_ + d];
  const float scale = 0.17677669529663687f;   // 1/sqrt(32)
  float mx = -3.4e38f;
  for (int s = 0; s < S; ++s) {
    float a = 0.f;
#pragma unroll
    for (int d = 0; d < 32; ++d) a += qv[d] * Ks[s*32+d];
    mx = fmaxf(mx, a * scale);
  }
  float sum = 0.f;
  float o[32];
#pragma unroll
  for (int d = 0; d < 32; ++d) o[d] = 0.f;
  for (int s = 0; s < S; ++s) {
    float a = 0.f;
#pragma unroll
    for (int d = 0; d < 32; ++d) a += qv[d] * Ks[s*32+d];
    float w = expf(a * scale - mx);
    sum += w;
#pragma unroll
    for (int d = 0; d < 32; ++d) o[d] += w * Vs[s*32+d];
  }
  float inv = 1.f / sum;
#pragma unroll
  for (int d = 0; d < 32; ++d) O[base + (size_t)l * DM_ + d] = o[d] * inv;
}

// residual + LayerNorm over DM=256, one wave per row, wave32 shuffle reduction
__global__ __launch_bounds__(32)
void resln256(const float* __restrict__ X, const float* __restrict__ Hh,
              const float* __restrict__ g, const float* __restrict__ b,
              float* __restrict__ Out, const int* __restrict__ active, int eid)
{
  if (active && active[eid] == 0) return;
  const size_t row = blockIdx.x;
  const int lane = threadIdx.x;
  float v[8];
  float s = 0.f;
#pragma unroll
  for (int j = 0; j < 8; ++j) {
    size_t idx = row * DM_ + lane * 8 + j;
    float t = X[idx] + Hh[idx];
    v[j] = t; s += t;
  }
#pragma unroll
  for (int off = 16; off > 0; off >>= 1) s += __shfl_xor(s, off, 32);
  float mean = s * (1.f / 256.f);
  float var = 0.f;
#pragma unroll
  for (int j = 0; j < 8; ++j) { float d = v[j] - mean; var += d * d; }
#pragma unroll
  for (int off = 16; off > 0; off >>= 1) var += __shfl_xor(var, off, 32);
  var *= (1.f / 256.f);
  float inv = 1.f / sqrtf(var + 1e-5f);
#pragma unroll
  for (int j = 0; j < 8; ++j) {
    int c = lane * 8 + j;
    Out[row * DM_ + c] = (v[j] - mean) * inv * g[c] + b[c];
  }
}

// residual + LayerNorm over D=16 (main ln3), one thread per row
__global__ void resln16(const float* __restrict__ A, const float* __restrict__ Y,
                        const float* __restrict__ g, const float* __restrict__ b,
                        float* __restrict__ Out, int rows)
{
  int r = blockIdx.x * blockDim.x + threadIdx.x;
  if (r >= rows) return;
  float v[16]; float m = 0.f;
#pragma unroll
  for (int d = 0; d < 16; ++d) { float t = A[(size_t)r*16+d] + Y[(size_t)r*16+d]; v[d]=t; m+=t; }
  m *= (1.f/16.f);
  float var = 0.f;
#pragma unroll
  for (int d = 0; d < 16; ++d) { float t = v[d]-m; var += t*t; }
  var *= (1.f/16.f);
  float inv = 1.f / sqrtf(var + 1e-5f);
#pragma unroll
  for (int d = 0; d < 16; ++d) Out[(size_t)r*16+d] = (v[d]-m)*inv*g[d] + b[d];
}

// x2 [bc,l,d] -> xT [(bc*16+d), l]   (shared input to all expert "start" GEMMs)
__global__ void build_xT(const float* __restrict__ x2, float* __restrict__ xT)
{
  int i = blockIdx.x * blockDim.x + threadIdx.x;
  if (i >= XELEMS) return;
  int d = i & 15; int l = (i >> 4) % L_; int bc = i / (L_ * D_);
  xT[((size_t)bc * 16 + d) * L_ + l] = x2[i];
}

// hT [(bc*16+d), Lp] -> tok [(bc*F+f), ip*16+d]
__global__ void build_tok(const float* __restrict__ hT, float* __restrict__ tok,
                          int p, int F, int Lp, const int* __restrict__ active, int eid)
{
  if (active && active[eid] == 0) return;
  int i = blockIdx.x * blockDim.x + threadIdx.x;
  int total = 2048 * Lp;
  if (i >= total) return;
  int d = i & 15; int rest = i >> 4;
  int ip = rest % p; rest /= p;
  int f = rest % F; int bc = rest / F;
  tok[i] = hT[((size_t)bc * 16 + d) * Lp + (size_t)f * p + ip];
}

// ---------------- router ----------------
__global__ void router_z(const float* __restrict__ x2, const float* __restrict__ sw,
                         const float* __restrict__ sb, float* __restrict__ z)
{
  int i = blockIdx.x * blockDim.x + threadIdx.x;
  if (i >= B_ * L_) return;
  int b = i / L_, l = i % L_;
  float acc = 0.f;
  for (int c = 0; c < C_; ++c) {
    const float* xp = x2 + (((size_t)(b * C_ + c)) * L_ + l) * D_;
    float sd = 0.f;
#pragma unroll
    for (int d = 0; d < 16; ++d) sd += xp[d];
    acc += sd * sw[c];
  }
  z[i] = acc * (1.f / 16.f) + sb[0];
}

__global__ void router_fft(const float* __restrict__ z, float* __restrict__ xr,
                           float* __restrict__ xi)
{
  int i = blockIdx.x * blockDim.x + threadIdx.x;
  if (i >= B_ * NF_) return;
  int b = i / NF_; int k = (i % NF_) + 1;
  float sr = 0.f, si = 0.f;
  for (int l = 0; l < L_; ++l) {
    float ang = -6.2831853071795864769f * (float)(k * l) / 96.f;
    float zv = z[b * L_ + l];
    sr += zv * cosf(ang);
    si += zv * sinf(ang);
  }
  const float scale = 0.10206207261596575f;  // 1/sqrt(96), norm='ortho'
  xr[i] = sr * scale;
  xi[i] = si * scale;
}

__global__ void router_l1(const float* __restrict__ xr, const float* __restrict__ xi,
                          const float* __restrict__ w1, const float* __restrict__ b1,
                          float* __restrict__ o1r, float* __restrict__ o1i)
{
  int i = blockIdx.x * blockDim.x + threadIdx.x;
  if (i >= B_ * 4 * NF_) return;
  int b = i / (4 * NF_), j = i % (4 * NF_);
  const float* w1r = w1;
  const float* w1i = w1 + NF_ * 4 * NF_;
  float ar = 0.f, ai = 0.f;
  for (int t = 0; t < NF_; ++t) {
    float r = xr[b * NF_ + t], im = xi[b * NF_ + t];
    float wr = w1r[t * (4 * NF_) + j], wi = w1i[t * (4 * NF_) + j];
    ar += r * wr - im * wi;
    ai += im * wr + r * wi;
  }
  o1r[i] = fmaxf(ar + b1[j], 0.f);
  o1i[i] = fmaxf(ai + b1[4 * NF_ + j], 0.f);
}

__global__ void router_l2(const float* __restrict__ o1r, const float* __restrict__ o1i,
                          const float* __restrict__ w2, const float* __restrict__ b2,
                          float* __restrict__ amp)
{
  int i = blockIdx.x * blockDim.x + threadIdx.x;
  if (i >= B_ * NF_) return;
  int b = i / NF_, j = i % NF_;
  const float* w2r = w2;
  const float* w2i = w2 + 4 * NF_ * NF_;
  float ar = b2[j], ai = b2[NF_ + j];
  for (int t = 0; t < 4 * NF_; ++t) {
    float r = o1r[b * 4 * NF_ + t], im = o1i[b * 4 * NF_ + t];
    float wr = w2r[t * NF_ + j], wi = w2i[t * NF_ + j];
    ar += r * wr - im * wi;
    ai += im * wr + r * wi;
  }
  amp[i] = sqrtf(ar * ar + ai * ai);
}

// logits, top-5 selection (first-index tie-break like lax.top_k), softmax, flags
__global__ void router_gate(const float* __restrict__ amp, const float* __restrict__ wg,
                            float* __restrict__ gates, int* __restrict__ active)
{
  if (threadIdx.x != 0 || blockIdx.x != 0) return;
  for (int e = 0; e < NE_; ++e) active[e] = 0;
  for (int b = 0; b < B_; ++b) {
    float lg[NE_];
    for (int e = 0; e < NE_; ++e) {
      float s = 0.f;
      for (int j = 0; j < NF_; ++j) s += amp[b * NF_ + j] * wg[j * NE_ + e];
      lg[e] = s;
    }
    int sel[5]; float val[5];
    unsigned used = 0;
    for (int t = 0; t < 5; ++t) {
      int best = 0; float bv = -3.4e38f;
      for (int e = 0; e < NE_; ++e)
        if (!((used >> e) & 1u) && lg[e] > bv) { bv = lg[e]; best = e; }
      used |= (1u << best); sel[t] = best; val[t] = bv;
    }
    float mx = val[0];
    float s = 0.f;
    for (int t = 0; t < 5; ++t) { val[t] = expf(val[t] - mx); s += val[t]; }
    for (int e = 0; e < NE_; ++e) gates[b * NE_ + e] = 0.f;
    for (int t = 0; t < 5; ++t) { gates[b * NE_ + sel[t]] = val[t] / s; active[sel[t]] = 1; }
  }
}

// per-expert: comb += gate * exp(LN_16(x2 + up_out))   (experts run sequentially)
__global__ void combine_expert(const float* __restrict__ X2, const float* __restrict__ U,
                               const float* __restrict__ g, const float* __restrict__ b,
                               const float* __restrict__ gates, float* __restrict__ comb,
                               int p, int F, int eid, const int* __restrict__ active)
{
  if (active && active[eid] == 0) return;
  int i = blockIdx.x * blockDim.x + threadIdx.x;
  if (i >= B_ * C_ * L_) return;
  int l = i % L_; int bc = i / L_; int bb = bc / C_;
  float gate = gates[bb * NE_ + eid];
  if (gate == 0.f) return;
  int f = l / p, ip = l - f * p;
  const float* ur = U + ((size_t)(bc * F + f) * p + ip) * 16;
  const float* xr = X2 + (size_t)i * 16;
  float v[16]; float m = 0.f;
#pragma unroll
  for (int d = 0; d < 16; ++d) { float t = xr[d] + ur[d]; v[d] = t; m += t; }
  m *= (1.f / 16.f);
  float var = 0.f;
#pragma unroll
  for (int d = 0; d < 16; ++d) { float t = v[d] - m; var += t * t; }
  var *= (1.f / 16.f);
  float inv = 1.f / sqrtf(var + 1e-5f);
#pragma unroll
  for (int d = 0; d < 16; ++d) {
    float t = (v[d] - m) * inv * g[d] + b[d];
    comb[(size_t)i * 16 + d] += gate * expf(t);
  }
}

__global__ void finalize_log(float* __restrict__ out, int n)
{
  int i = blockIdx.x * blockDim.x + threadIdx.x;
  if (i >= n) return;
  float v = out[i];
  out[i] = logf(v == 0.f ? 2.220446049250313e-16f : v);
}

// ---------------------------------------------------------------------------
// Parameter pytree indexing
// ---------------------------------------------------------------------------
struct Lin    { const float* w; const float* b; };
struct AttnP  { Lin q, k, v, o; };
struct LNP    { const float* g; const float* b; };
struct ExpertP{ Lin start, down, fc1, fc2, up; AttnP at; LNP ln1, ln2, ln3; const float* pe; };
struct RouterP{ Lin start; const float *w1, *b1, *w2, *b2, *wg; };
struct NetP {
  Lin down, fc1, fc2, up; AttnP at; LNP ln1, ln2, ln3; const float* pe;
  RouterP r; ExpertP ex[NE_]; const float* x;
};

// JAX tree_leaves order: dict keys sorted alphabetically at every level.
static void fill_sorted(void* const* d_in, NetP& n)
{
  int t = 0;
  auto nx = [&]() { return (const float*)d_in[t++]; };
  auto lin_bw = [&](Lin& L) { L.b = nx(); L.w = nx(); };
  auto attn_s = [&](AttnP& a) { lin_bw(a.k); lin_bw(a.o); lin_bw(a.q); lin_bw(a.v); };
  auto ln_bg  = [&](LNP& l) { l.b = nx(); l.g = nx(); };
  attn_s(n.at);
  lin_bw(n.down);
  for (int e = 0; e < NE_; ++e) {
    ExpertP& E = n.ex[e];
    attn_s(E.at); lin_bw(E.down); lin_bw(E.fc1); lin_bw(E.fc2);
    ln_bg(E.ln1); ln_bg(E.ln2); ln_bg(E.ln3);
    E.pe = nx(); lin_bw(E.start); lin_bw(E.up);
  }
  lin_bw(n.fc1); lin_bw(n.fc2); ln_bg(n.ln1); ln_bg(n.ln2); ln_bg(n.ln3);
  n.pe = nx();
  n.r.b1 = nx(); n.r.b2 = nx(); lin_bw(n.r.start);
  n.r.w1 = nx(); n.r.w2 = nx(); n.r.wg = nx();
  lin_bw(n.up);
  n.x = nx();
}

// Insertion-order recursive flatten (dict insertion order as written in reference)
static void fill_insertion(void* const* d_in, NetP& n)
{
  int t = 0;
  auto nx = [&]() { return (const float*)d_in[t++]; };
  auto lin_wb = [&](Lin& L) { L.w = nx(); L.b = nx(); };
  auto attn_i = [&](AttnP& a) { lin_wb(a.q); lin_wb(a.k); lin_wb(a.v); lin_wb(a.o); };
  auto ln_gb  = [&](LNP& l) { l.g = nx(); l.b = nx(); };
  n.x = nx();
  lin_wb(n.down); attn_i(n.at); lin_wb(n.fc1); lin_wb(n.fc2); lin_wb(n.up);
  ln_gb(n.ln1); ln_gb(n.ln2); ln_gb(n.ln3); n.pe = nx();
  lin_wb(n.r.start);
  n.r.w1 = nx(); n.r.b1 = nx(); n.r.w2 = nx(); n.r.b2 = nx(); n.r.wg = nx();
  for (int e = 0; e < NE_; ++e) {
    ExpertP& E = n.ex[e];
    lin_wb(E.start); lin_wb(E.down); attn_i(E.at);
    lin_wb(E.fc1); lin_wb(E.fc2); lin_wb(E.up);
    ln_gb(E.ln1); ln_gb(E.ln2); ln_gb(E.ln3); E.pe = nx();
  }
}

// ---------------------------------------------------------------------------
extern "C" void kernel_launch(void* const* d_in, const int* in_sizes, int n_in,
                              void* d_out, int out_size, void* d_ws, size_t ws_size,
                              hipStream_t stream)
{
  (void)ws_size;
  NetP net;
  if (n_in > 0 && in_sizes[0] == XELEMS) fill_insertion(d_in, net);
  else                                   fill_sorted(d_in, net);

  float* ws = (float*)d_ws;
  size_t off = 0;
  auto alloc = [&](size_t nfl) { float* p = ws + off; off += nfl; return p; };
  const int MAXTOK = 6144;                       // 128 sequences * max F (48)
  float* bH   = alloc((size_t)MAXTOK * DM_);
  float* bQ   = alloc((size_t)MAXTOK * DM_);
  float* bK   = alloc((size_t)MAXTOK * DM_);
  float* bV   = alloc((size_t)MAXTOK * DM_);
  float* bAO  = alloc((size_t)MAXTOK * DM_);
  float* bT   = alloc((size_t)MAXTOK * DM_);
  float* bFF  = alloc((size_t)MAXTOK * DFF_);
  float* bX2  = alloc(XELEMS);
  float* bXT  = alloc(XELEMS);
  float* bHT  = alloc((size_t)2048 * 120);       // max Lp = 114
  float* bTok = alloc((size_t)2048 * 120);       // tok / up-out / main-up reuse
  float* bZ   = alloc(B_ * L_);
  float* bXr  = alloc(B_ * NF_);
  float* bXi  = alloc(B_ * NF_);
  float* bO1r = alloc(B_ * 4 * NF_);
  float* bO1i = alloc(B_ * 4 * NF_);
  float* bAmp = alloc(B_ * NF_);
  float* bGates = alloc(B_ * NE_);
  int*   bAct = (int*)alloc(32);

  float* out = (float*)d_out;

  auto gemm = [&](const float* A, Lin w, const float* pe, int pe_mod, float* Cp,
                  int M, int N, int Kd, int relu, const int* act, int e) {
    dim3 grid((N + 63) / 64, M / 32);            // M always multiple of 32 here
    gemm_wmma<<<grid, 128, 0, stream>>>(A, w.w, w.b, pe, pe_mod, Cp, M, N, Kd, relu, act, e);
  };
  auto run_block = [&](float* x, int ntok, int S, int nseq, const AttnP& at,
                       Lin fc1, Lin fc2, const LNP& ln1, const LNP& ln2,
                       const int* act, int e) {
    gemm(x, at.q, nullptr, 1, bQ, ntok, DM_, DM_, 0, act, e);
    gemm(x, at.k, nullptr, 1, bK, ntok, DM_, DM_, 0, act, e);
    gemm(x, at.v, nullptr, 1, bV, ntok, DM_, DM_, 0, act, e);
    attn_fused<<<dim3(nseq, H_), S, (size_t)S * 64 * sizeof(float), stream>>>(
        bQ, bK, bV, bAO, S, act, e);
    gemm(bAO, at.o, nullptr, 1, bT, ntok, DM_, DM_, 0, act, e);
    resln256<<<ntok, 32, 0, stream>>>(x, bT, ln1.g, ln1.b, bQ, act, e);
    gemm(bQ, fc1, nullptr, 1, bFF, ntok, DFF_, DM_, 1, act, e);
    gemm(bFF, fc2, nullptr, 1, bK, ntok, DM_, DFF_, 0, act, e);
    resln256<<<ntok, 32, 0, stream>>>(bQ, bK, ln2.g, ln2.b, x, act, e);
  };

  // ---------------- Stage 1 (main transformer) ----------------
  gemm(net.x, net.down, net.pe, C_, bH, NTOK_MAIN, DM_, L_ * D_, 0, nullptr, 0);
  run_block(bH, NTOK_MAIN, C_, B_, net.at, net.fc1, net.fc2, net.ln1, net.ln2, nullptr, 0);
  gemm(bH, net.up, nullptr, 1, bTok, NTOK_MAIN, L_ * D_, DM_, 0, nullptr, 0);
  resln16<<<(NROWS16 + 255) / 256, 256, 0, stream>>>(net.x, bTok, net.ln3.g, net.ln3.b,
                                                     bX2, NROWS16);

  // ---------------- Router (DFT + complex MLP + top-5) ----------------
  router_z  <<<2, 256, 0, stream>>>(bX2, net.r.start.w, net.r.start.b, bZ);
  router_fft<<<1, 256, 0, stream>>>(bZ, bXr, bXi);
  router_l1 <<<3, 256, 0, stream>>>(bXr, bXi, net.r.w1, net.r.b1, bO1r, bO1i);
  router_l2 <<<1, 256, 0, stream>>>(bO1r, bO1i, net.r.w2, net.r.b2, bAmp);
  router_gate<<<1, 32, 0, stream>>>(bAmp, net.r.wg, bGates, bAct);

  hipMemsetAsync(d_out, 0, (size_t)out_size * sizeof(float), stream);

  // ---------------- Experts (device-side gated) ----------------
  build_xT<<<(XELEMS + 255) / 256, 256, 0, stream>>>(bX2, bXT);
  static const int PS[NE_] = {96, 48, 32, 24, 19, 16, 13, 12, 10, 9, 8, 7, 6, 5, 4, 3, 2};
  for (int e = 0; e < NE_; ++e) {
    int p = PS[e], F = (L_ + p - 1) / p, Lp = F * p, ntok = NTOK_MAIN * F;
    const ExpertP& E = net.ex[e];
    gemm(bXT, E.start, nullptr, 1, bHT, 2048, Lp, L_, 0, bAct, e);
    int tot = 2048 * Lp;
    build_tok<<<(tot + 255) / 256, 256, 0, stream>>>(bHT, bTok, p, F, Lp, bAct, e);
    gemm(bTok, E.down, E.pe, F, bH, ntok, DM_, p * D_, 0, bAct, e);
    run_block(bH, ntok, F, NTOK_MAIN, E.at, E.fc1, E.fc2, E.ln1, E.ln2, bAct, e);
    gemm(bH, E.up, nullptr, 1, bTok, ntok, p * D_, DM_, 0, bAct, e);
    combine_expert<<<(B_ * C_ * L_ + 255) / 256, 256, 0, stream>>>(
        bX2, bTok, E.ln3.g, E.ln3.b, bGates, out, p, F, e, bAct);
  }

  finalize_log<<<(out_size + 255) / 256, 256, 0, stream>>>(out, out_size);
}
